// Attention_27754078667084
// MI455X (gfx1250) — matrix operand order repaired and checked by reading, not compiled, not run on hardware
//
#include <hip/hip_runtime.h>

typedef __bf16 bf16_t;
typedef bf16_t v16bf __attribute__((ext_vector_type(16)));
typedef bf16_t v8bf  __attribute__((ext_vector_type(8)));
typedef float  v8f   __attribute__((ext_vector_type(8)));

#define NUM_B  8
#define NUM_N  1024
#define NUM_NQ 256
#define NUM_C  768
#define NUM_L  1280   /* N + Nq */
#define NUM_H  12
#define HEAD_D 64

// ---------------------------------------------------------------------------
// Fragment helpers per CDNA5 ISA 7.12.2 (wave32):
//   A (16x32, 16-bit):  lane l -> row M = l&15 ; elements e<8 -> K = 8*hi+e,
//                       e>=8 -> K = 16+8*hi+(e-8), hi = l>>4
//   B (32x16, 16-bit):  lane l -> col N = l&15 ; same K mapping as A
//   C/D (16x16, f32) :  VGPR r, lane l -> M = r+8*hi, N = l&15
// ---------------------------------------------------------------------------
static __device__ __forceinline__ v16bf load_a_frag(const bf16_t* rowbase, int hi) {
  v8bf lo = *reinterpret_cast<const v8bf*>(rowbase + 8 * hi);
  v8bf up = *reinterpret_cast<const v8bf*>(rowbase + 16 + 8 * hi);
  return __builtin_shufflevector(lo, up, 0,1,2,3,4,5,6,7,8,9,10,11,12,13,14,15);
}

static __device__ __forceinline__ v16bf load_b_frag(const bf16_t* colrow, int hi) {
  v8bf lo = *reinterpret_cast<const v8bf*>(colrow + 8 * hi);
  v8bf up = *reinterpret_cast<const v8bf*>(colrow + 16 + 8 * hi);
  return __builtin_shufflevector(lo, up, 0,1,2,3,4,5,6,7,8,9,10,11,12,13,14,15);
}

static __device__ __forceinline__ v8f wmma_bf16(v16bf a, v16bf b, v8f c) {
  return __builtin_amdgcn_wmma_f32_16x16x32_bf16(false, a, false, b, (short)0, c,
                                                 false, false);
}

// ---------------------------------------------------------------------------
// K1: concat [x ; q] -> bf16 workspace  (xs[b, l, c])
// ---------------------------------------------------------------------------
__global__ void build_xs_kernel(const float* __restrict__ x, const float* __restrict__ q,
                                bf16_t* __restrict__ xs, int total) {
  int i = blockIdx.x * blockDim.x + threadIdx.x;
  if (i >= total) return;
  int c  = i % NUM_C;
  int rl = (i / NUM_C) % NUM_L;
  int b  = i / (NUM_C * NUM_L);
  float v = (rl < NUM_N) ? x[((size_t)b * NUM_N + rl) * NUM_C + c]
                         : q[((size_t)b * NUM_NQ + (rl - NUM_N)) * NUM_C + c];
  xs[i] = (bf16_t)v;
}

// K2: f32 -> bf16 weight convert
__global__ void cvt_kernel(const float* __restrict__ s, bf16_t* __restrict__ d, int n) {
  int i = blockIdx.x * blockDim.x + threadIdx.x;
  if (i < n) d[i] = (bf16_t)s[i];
}

// ---------------------------------------------------------------------------
// K3: QKV projection.  P = xs @ Wqkv^T  (rows = B*L, cols = 3C)
// one wave -> 32 rows x 64 cols: each B fragment feeds two WMMAs (2 M-tiles),
// halving L2 weight traffic vs 16-row tiles.  K stepped by 32 (8 wmma/step).
// Epilogue scatters into Q (scaled by d^-0.5), K [B,h,L,d], V^T [B,h,d,L].
// ---------------------------------------------------------------------------
__global__ __launch_bounds__(32)
void qkv_gemm_kernel(const bf16_t* __restrict__ xs, const bf16_t* __restrict__ Wq,
                     bf16_t* __restrict__ Qs, bf16_t* __restrict__ Kt,
                     bf16_t* __restrict__ Vt) {
  const int lane = threadIdx.x & 31;
  const int ln = lane & 15, hi = lane >> 4;
  const int row0 = blockIdx.x * 32;
  const int col0 = blockIdx.y * 64;

  v8f acc[2][4] = {};
  const bf16_t* arow0 = xs + (size_t)(row0 + ln) * NUM_C;
  const bf16_t* arow1 = xs + (size_t)(row0 + 16 + ln) * NUM_C;

  for (int kb = 0; kb < NUM_C; kb += 32) {
    v16bf a0 = load_a_frag(arow0 + kb, hi);
    v16bf a1 = load_a_frag(arow1 + kb, hi);
#pragma unroll
    for (int t = 0; t < 4; ++t) {
      v16bf bf = load_b_frag(Wq + (size_t)(col0 + 16 * t + ln) * NUM_C + kb, hi);
      acc[0][t] = wmma_bf16(a0, bf, acc[0][t]);
      acc[1][t] = wmma_bf16(a1, bf, acc[1][t]);
    }
  }

  const int which = col0 / NUM_C;              // 0:Q 1:K 2:V
  const int head  = (col0 % NUM_C) / HEAD_D;   // 64-col block == one head
#pragma unroll
  for (int mt = 0; mt < 2; ++mt) {
#pragma unroll
    for (int t = 0; t < 4; ++t) {
      const int dd = 16 * t + ln;
#pragma unroll
      for (int r = 0; r < 8; ++r) {
        int m = row0 + 16 * mt + r + 8 * hi;
        int bb = m / NUM_L, l = m % NUM_L;
        float v = acc[mt][t][r];
        size_t bh = (size_t)bb * NUM_H + head;
        if (which == 0)
          Qs[(bh * NUM_L + l) * HEAD_D + dd] = (bf16_t)(v * 0.125f);  // * d^-0.5
        else if (which == 1)
          Kt[(bh * NUM_L + l) * HEAD_D + dd] = (bf16_t)v;
        else
          Vt[(bh * HEAD_D + dd) * NUM_L + l] = (bf16_t)v;
      }
    }
  }
}

// ---------------------------------------------------------------------------
// K4: flash attention. One wave handles 32 query rows (two 16-row M-tiles)
// for one (b, head), streaming 32 keys per step: 8 wmma for S (K fragments
// reused by both M-tiles), dual online softmax, P -> LDS -> A-fragments,
// 8 wmma for P@V (V fragments reused by both M-tiles).
// ---------------------------------------------------------------------------
__global__ __launch_bounds__(32)
void attn_kernel(const bf16_t* __restrict__ Qs, const bf16_t* __restrict__ Kt,
                 const bf16_t* __restrict__ Vt, const int* __restrict__ qmask,
                 bf16_t* __restrict__ Obf) {
  __shared__ __align__(64) bf16_t Pt[2][16 * 32];

  const int lane = threadIdx.x & 31, ln = lane & 15, hi = lane >> 4;
  const int bh = blockIdx.y, b = bh / NUM_H, head = bh % NUM_H;
  const int q0 = blockIdx.x * 32;

  const bf16_t* Qp = Qs + (size_t)bh * NUM_L * HEAD_D;
  const bf16_t* Kp = Kt + (size_t)bh * NUM_L * HEAD_D;
  const bf16_t* Vp = Vt + (size_t)bh * HEAD_D * NUM_L;

  v16bf aq[2][2];   // [M-tile][d-chunk]
#pragma unroll
  for (int mt = 0; mt < 2; ++mt) {
    const bf16_t* qrow = Qp + (size_t)(q0 + 16 * mt + ln) * HEAD_D;
    aq[mt][0] = load_a_frag(qrow, hi);        // d-chunk 0..31
    aq[mt][1] = load_a_frag(qrow + 32, hi);   // d-chunk 32..63
  }

  v8f o[2][4] = {};
  float mrow[2][8], lsum[2][8];
#pragma unroll
  for (int mt = 0; mt < 2; ++mt)
#pragma unroll
    for (int r = 0; r < 8; ++r) { mrow[mt][r] = -1e30f; lsum[mt][r] = 0.0f; }

  for (int j0 = 0; j0 < NUM_L; j0 += 32) {
    v8f s[2][2] = {};   // [M-tile][key-tile]
#pragma unroll
    for (int t = 0; t < 2; ++t) {
      const int key = j0 + 16 * t + ln;          // this lane's key column
      const bf16_t* krow = Kp + (size_t)key * HEAD_D;
      v16bf kb0 = load_b_frag(krow, hi);
      v16bf kb1 = load_b_frag(krow + 32, hi);
#pragma unroll
      for (int mt = 0; mt < 2; ++mt) {
        s[mt][t] = wmma_bf16(aq[mt][0], kb0, s[mt][t]);
        s[mt][t] = wmma_bf16(aq[mt][1], kb1, s[mt][t]);
      }
      const bool valid =
          (key < NUM_N) || (qmask[b * NUM_NQ + (key - NUM_N)] != 0);
      if (!valid) {
#pragma unroll
        for (int mt = 0; mt < 2; ++mt)
#pragma unroll
          for (int r = 0; r < 8; ++r) s[mt][t][r] = -1e30f;
      }
    }

    // online softmax over the 32 new key columns (rows live in 16-lane halves)
#pragma unroll
    for (int mt = 0; mt < 2; ++mt) {
#pragma unroll
      for (int r = 0; r < 8; ++r) {
        float mx = fmaxf(s[mt][0][r], s[mt][1][r]);
#pragma unroll
        for (int off = 1; off < 16; off <<= 1)
          mx = fmaxf(mx, __shfl_xor(mx, off, 32));
        float nm   = fmaxf(mrow[mt][r], mx);
        float corr = __expf(mrow[mt][r] - nm);
        mrow[mt][r] = nm;
        float p0 = __expf(s[mt][0][r] - nm);
        float p1 = __expf(s[mt][1][r] - nm);
        s[mt][0][r] = p0; s[mt][1][r] = p1;
        float rs = p0 + p1;
#pragma unroll
        for (int off = 1; off < 16; off <<= 1)
          rs += __shfl_xor(rs, off, 32);
        lsum[mt][r] = lsum[mt][r] * corr + rs;
#pragma unroll
        for (int nt = 0; nt < 4; ++nt) o[mt][nt][r] *= corr;
      }

      // D-layout -> row-major LDS tile [16 rows][32 keys] (bf16)
#pragma unroll
      for (int t = 0; t < 2; ++t)
#pragma unroll
        for (int r = 0; r < 8; ++r)
          Pt[mt][(r + 8 * hi) * 32 + 16 * t + ln] = (bf16_t)s[mt][t][r];
    }
    __builtin_amdgcn_wave_barrier();   // DS ops are in-order within a wave

    v16bf pa0 = load_a_frag(&Pt[0][ln * 32], hi);   // A-fragment of P (16x32)
    v16bf pa1 = load_a_frag(&Pt[1][ln * 32], hi);
    __builtin_amdgcn_wave_barrier();

#pragma unroll
    for (int nt = 0; nt < 4; ++nt) {
      const bf16_t* vrow = Vp + (size_t)(16 * nt + ln) * NUM_L + j0;
      v16bf vb = load_b_frag(vrow, hi);
      o[0][nt] = wmma_bf16(pa0, vb, o[0][nt]);
      o[1][nt] = wmma_bf16(pa1, vb, o[1][nt]);
    }
  }

  // normalize and write O[b, l, head*64 + dd] (bf16)
#pragma unroll
  for (int mt = 0; mt < 2; ++mt) {
#pragma unroll
    for (int r = 0; r < 8; ++r) {
      float inv = 1.0f / lsum[mt][r];
      int m = q0 + 16 * mt + r + 8 * hi;
      bf16_t* orow = Obf + ((size_t)b * NUM_L + m) * NUM_C + head * HEAD_D;
#pragma unroll
      for (int nt = 0; nt < 4; ++nt)
        orow[16 * nt + ln] = (bf16_t)(o[mt][nt][r] * inv);
    }
  }
}

// ---------------------------------------------------------------------------
// K5: output projection out = O @ Wo^T + bo, written with the tuple split:
// first all out[:, :N, :], then all out[:, N:, :].
// One wave -> 32 rows x 64 cols (B fragments reused by both M-tiles).
// ---------------------------------------------------------------------------
__global__ __launch_bounds__(32)
void proj_gemm_kernel(const bf16_t* __restrict__ Obf, const bf16_t* __restrict__ Wob,
                      const float* __restrict__ bo, float* __restrict__ out) {
  const int lane = threadIdx.x & 31;
  const int ln = lane & 15, hi = lane >> 4;
  const int row0 = blockIdx.x * 32;
  const int col0 = blockIdx.y * 64;

  v8f acc[2][4] = {};
  const bf16_t* arow0 = Obf + (size_t)(row0 + ln) * NUM_C;
  const bf16_t* arow1 = Obf + (size_t)(row0 + 16 + ln) * NUM_C;

  for (int kb = 0; kb < NUM_C; kb += 32) {
    v16bf a0 = load_a_frag(arow0 + kb, hi);
    v16bf a1 = load_a_frag(arow1 + kb, hi);
#pragma unroll
    for (int t = 0; t < 4; ++t) {
      v16bf bf = load_b_frag(Wob + (size_t)(col0 + 16 * t + ln) * NUM_C + kb, hi);
      acc[0][t] = wmma_bf16(a0, bf, acc[0][t]);
      acc[1][t] = wmma_bf16(a1, bf, acc[1][t]);
    }
  }

#pragma unroll
  for (int mt = 0; mt < 2; ++mt) {
#pragma unroll
    for (int t = 0; t < 4; ++t) {
      const int n = col0 + 16 * t + ln;
      const float bias = bo[n];
#pragma unroll
      for (int r = 0; r < 8; ++r) {
        int m = row0 + 16 * mt + r + 8 * hi;
        int bb = m / NUM_L, l = m % NUM_L;
        size_t dst = (l < NUM_N)
            ? ((size_t)bb * NUM_N + l) * NUM_C + n
            : (size_t)NUM_B * NUM_N * NUM_C + ((size_t)bb * NUM_NQ + (l - NUM_N)) * NUM_C + n;
        out[dst] = acc[mt][t][r] + bias;
      }
    }
  }
}

// ---------------------------------------------------------------------------
extern "C" void kernel_launch(void* const* d_in, const int* in_sizes, int n_in,
                              void* d_out, int out_size, void* d_ws, size_t ws_size,
                              hipStream_t stream) {
  (void)in_sizes; (void)n_in; (void)out_size; (void)ws_size;

  const float* x    = (const float*)d_in[0];
  const float* q    = (const float*)d_in[1];
  const int*   qm   = (const int*)d_in[2];
  const float* Wqkv = (const float*)d_in[3];
  const float* Wo   = (const float*)d_in[4];
  const float* bo   = (const float*)d_in[5];
  float* out = (float*)d_out;

  char* ws = (char*)d_ws;
  size_t off = 0;
  auto alloc = [&](size_t bytes) {
    char* p = ws + off;
    off += (bytes + 255) & ~(size_t)255;
    return p;
  };
  bf16_t* xs  = (bf16_t*)alloc((size_t)NUM_B * NUM_L * NUM_C * 2);
  bf16_t* Wqb = (bf16_t*)alloc((size_t)3 * NUM_C * NUM_C * 2);
  bf16_t* Wob = (bf16_t*)alloc((size_t)NUM_C * NUM_C * 2);
  bf16_t* Qs  = (bf16_t*)alloc((size_t)NUM_B * NUM_H * NUM_L * HEAD_D * 2);
  bf16_t* Kt  = (bf16_t*)alloc((size_t)NUM_B * NUM_H * NUM_L * HEAD_D * 2);
  bf16_t* Vt  = (bf16_t*)alloc((size_t)NUM_B * NUM_H * NUM_L * HEAD_D * 2);
  bf16_t* Obf = (bf16_t*)alloc((size_t)NUM_B * NUM_L * NUM_C * 2);

  const int nxs = NUM_B * NUM_L * NUM_C;
  build_xs_kernel<<<(nxs + 255) / 256, 256, 0, stream>>>(x, q, xs, nxs);
  const int nwq = 3 * NUM_C * NUM_C;
  cvt_kernel<<<(nwq + 255) / 256, 256, 0, stream>>>(Wqkv, Wqb, nwq);
  const int nwo = NUM_C * NUM_C;
  cvt_kernel<<<(nwo + 255) / 256, 256, 0, stream>>>(Wo, Wob, nwo);

  dim3 g1(NUM_B * NUM_L / 32, 3 * NUM_C / 64);
  qkv_gemm_kernel<<<g1, 32, 0, stream>>>(xs, Wqb, Qs, Kt, Vt);

  dim3 g2(NUM_L / 32, NUM_B * NUM_H);
  attn_kernel<<<g2, 32, 0, stream>>>(Qs, Kt, Vt, qm, Obf);

  dim3 g3(NUM_B * NUM_L / 32, NUM_C / 64);
  proj_gemm_kernel<<<g3, 32, 0, stream>>>(Obf, Wob, bo, out);
}
